// SubpixelCorrelationWrapper_15324443312760
// MI455X (gfx1250) — matrix operand order repaired and checked by reading, not compiled, not run on hardware
//
#include <hip/hip_runtime.h>
#include <hip/hip_bf16.h>

typedef __attribute__((ext_vector_type(16))) _Float16 v16h;
typedef __attribute__((ext_vector_type(8)))  _Float16 v8h;
typedef __attribute__((ext_vector_type(8)))  float    v8f;

#define Bsz  2
#define Cin  256
#define Hh   120
#define Ww   160
#define Oc   32
#define Dn   64
#define HWs  (Hh * Ww)      // 19200
#define ROWS (Bsz * Hh)     // 240

// ---------------------------------------------------------------------------
// Kernel 1: channel projection.
//   proj[row][w][o] (f16) = sum_c w_down[o][c] * in[b][c][h][w],  row = b*H+h
// One wave owns a 16-pixel x 32-output tile: 2 accumulators, 8 K-steps of
// v_wmma_f32_16x16x32_f16 each. Weights live in LDS as f16.
// ---------------------------------------------------------------------------
__global__ __launch_bounds__(256) void proj_kernel(
    const float* __restrict__ left, const float* __restrict__ right,
    const float* __restrict__ w_down,
    _Float16* __restrict__ lproj, _Float16* __restrict__ rproj)
{
    __shared__ _Float16 Wl[Oc * Cin];                 // 16 KB
    const int tid = threadIdx.x;
    for (int i = tid; i < Oc * Cin; i += 256) Wl[i] = (_Float16)w_down[i];
    __syncthreads();

    const float*    __restrict__ in   = (blockIdx.y == 0) ? left  : right;
    _Float16*       __restrict__ outp = (blockIdx.y == 0) ? lproj : rproj;

    const int wave = tid >> 5, lane = tid & 31;
    const int m  = lane & 15;          // A/B/D "row within half" lane id
    const int hi = lane >> 4;          // which K half this lane holds

    const int ptile = blockIdx.x * 8 + wave;          // 2400 tiles total
    const int p0    = ptile * 16;
    const int row   = p0 / Ww;                        // b*H + h
    const int w0    = p0 - row * Ww;                  // tile never straddles a row (160 % 16 == 0)
    const int b     = row / Hh;
    const int h     = row - b * Hh;

    v8f acc0 = {}; v8f acc1 = {};
    const float* src_base = in + (size_t)b * Cin * HWs + (size_t)h * Ww + w0 + m;

    #pragma unroll
    for (int ks = 0; ks < 8; ++ks) {
        // A = weight tile (16 o-rows x 32 c). ISA layout: lane<16 holds K {0..7,16..23},
        // lane>=16 holds K {8..15,24..31}; element e<8 -> K=klo+e, e>=8 -> K=klo+8+e.
        const int klo = ks * 32 + hi * 8;
        v8h lo0 = *(const v8h*)&Wl[(m     ) * Cin + klo];
        v8h hh0 = *(const v8h*)&Wl[(m     ) * Cin + klo + 16];
        v8h lo1 = *(const v8h*)&Wl[(m + 16) * Cin + klo];
        v8h hh1 = *(const v8h*)&Wl[(m + 16) * Cin + klo + 16];
        v16h a0 = __builtin_shufflevector(lo0, hh0, 0,1,2,3,4,5,6,7,8,9,10,11,12,13,14,15);
        v16h a1 = __builtin_shufflevector(lo1, hh1, 0,1,2,3,4,5,6,7,8,9,10,11,12,13,14,15);

        // B = input tile (32 c x 16 pixels). Lane column N = m, K = hi*8 + e (e<8), 16 + hi*8 + (e-8) (e>=8).
        // Each e: the 16 low lanes read 16 contiguous floats (64B coalesced).
        const int cbase = ks * 32 + hi * 8;
        const float* sp = src_base + (size_t)cbase * HWs;
        v16h bm;
        #pragma unroll
        for (int e = 0; e < 16; ++e) bm[e] = (_Float16)sp[(size_t)(e + (e >= 8 ? 8 : 0)) * HWs];

        acc0 = __builtin_amdgcn_wmma_f32_16x16x32_f16(false, a0, false, bm,
                                                      (short)0, acc0, false, false);
        acc1 = __builtin_amdgcn_wmma_f32_16x16x32_f16(false, a1, false, bm,
                                                      (short)0, acc1, false, false);
    }

    // D layout: acc[r] lives at (o = otile*16 + hi*8 + r, pixel = p0 + m).
    // The 8 consecutive o-values pack into one 16B store per accumulator.
    v8h o0, o1;
    #pragma unroll
    for (int r = 0; r < 8; ++r) { o0[r] = (_Float16)acc0[r]; o1[r] = (_Float16)acc1[r]; }
    _Float16* dst = outp + ((size_t)row * Ww + w0 + m) * Oc;
    *(v8h*)(dst + hi * 8)       = o0;
    *(v8h*)(dst + 16 + hi * 8)  = o1;
}

// ---------------------------------------------------------------------------
// Kernel 2: per (b,h) row, build M[w][x] = sum_c l[c,w] r[c,x] (160x160, LDS)
// with one 16x16x32 WMMA per tile, then emit
//   corr[d][w] = (1-wx)*m0*M[w][x0c] + wx*m1*M[w][x1c].
// LDS staging uses CDNA5 async loads (memory -> LDS, no VGPR bounce),
// tracked on ASYNCcnt.
// ---------------------------------------------------------------------------
__global__ __launch_bounds__(256) void corr_kernel(
    const _Float16* __restrict__ lproj, const _Float16* __restrict__ rproj,
    float* __restrict__ out)
{
    __shared__ _Float16 Lt[Ww * Oc];      // 10 KB, [w][c]
    __shared__ _Float16 Rt[Ww * Oc];      // 10 KB, [x][c]
    __shared__ float    Mm[Ww * Ww];      // 100 KB

    const int tid = threadIdx.x;
    const int bh  = blockIdx.x;

    // Stage both projected rows (fully contiguous 10 KB each) straight into
    // LDS with global_load_async_to_lds_b128 (GV mode: vdst = LDS byte addr,
    // vaddr = 64-bit global addr). Low 32 bits of a generic shared pointer
    // are the LDS byte address per the flat-aperture rule.
    {
        const char* lg = (const char*)(lproj + (size_t)bh * Ww * Oc);
        const char* rg = (const char*)(rproj + (size_t)bh * Ww * Oc);
        const unsigned ltBase = (unsigned)(uintptr_t)&Lt[0];
        const unsigned rtBase = (unsigned)(uintptr_t)&Rt[0];
        for (int i = tid; i < (Ww * Oc * 2) / 16; i += 256) {   // 640 x 16B chunks each
            unsigned long long gl = (unsigned long long)(uintptr_t)(lg + i * 16);
            unsigned long long gr = (unsigned long long)(uintptr_t)(rg + i * 16);
            unsigned dl = ltBase + i * 16;
            unsigned dr = rtBase + i * 16;
            asm volatile("global_load_async_to_lds_b128 %0, %1, off"
                         :: "v"(dl), "v"(gl) : "memory");
            asm volatile("global_load_async_to_lds_b128 %0, %1, off"
                         :: "v"(dr), "v"(gr) : "memory");
        }
        asm volatile("s_wait_asynccnt 0x0" ::: "memory");
    }
    __syncthreads();

    const int wave = tid >> 5, lane = tid & 31;
    const int m = lane & 15, hi = lane >> 4;

    // 100 tiles of M, one WMMA each (K = 32 channels exactly).
    for (int t = wave; t < 100; t += 8) {
        const int ti = t / 10;
        const int tj = t - ti * 10;
        v8h lo = *(const v8h*)&Lt[(16 * ti + m) * Oc + hi * 16];
        v8h hh = *(const v8h*)&Lt[(16 * ti + m) * Oc + hi * 16 + 8];
        v16h a = __builtin_shufflevector(lo, hh, 0,1,2,3,4,5,6,7,8,9,10,11,12,13,14,15);
        v16h bm = *(const v16h*)&Rt[(16 * tj + m) * Oc + hi * 16];  // K=hi*16+e, N=m
        v8f d = {};
        d = __builtin_amdgcn_wmma_f32_16x16x32_f16(false, a, false, bm,
                                                   (short)0, d, false, false);
        #pragma unroll
        for (int r = 0; r < 8; ++r)
            Mm[(16 * ti + hi * 8 + r) * Ww + 16 * tj + m] = d[r];
    }
    __syncthreads();

    // Disparity sampling: consecutive lanes -> consecutive w, LDS stride 161
    // dwords (odd) -> conflict-free; output stores fully coalesced.
    const int b = bh / Hh, h = bh - b * Hh;
    const float step = 64.0f / 63.0f;                 // linspace(0, 64, 64)
    for (int e = tid; e < Dn * Ww; e += 256) {
        const int d = e / Ww;
        const int w = e - d * Ww;
        const float xs  = (float)w - (float)d * step;
        const float x0f = floorf(xs);
        const int   x0  = (int)x0f;
        const int   x1  = x0 + 1;
        const float wx  = xs - x0f;
        const float w0f = (x0 >= 0 && x0 < Ww) ? (1.0f - wx) : 0.0f;
        const float w1f = (x1 >= 0 && x1 < Ww) ? wx : 0.0f;
        const int x0c = x0 < 0 ? 0 : (x0 > Ww - 1 ? Ww - 1 : x0);
        const int x1c = x1 < 0 ? 0 : (x1 > Ww - 1 ? Ww - 1 : x1);
        const float v = Mm[w * Ww + x0c] * w0f + Mm[w * Ww + x1c] * w1f;
        out[(((size_t)b * Dn + d) * Hh + h) * Ww + w] = v;
    }
}

// ---------------------------------------------------------------------------
extern "C" void kernel_launch(void* const* d_in, const int* in_sizes, int n_in,
                              void* d_out, int out_size, void* d_ws, size_t ws_size,
                              hipStream_t stream) {
    const float* left  = (const float*)d_in[0];   // (2,256,120,160)
    const float* right = (const float*)d_in[1];   // (2,256,120,160)
    const float* wdn   = (const float*)d_in[2];   // (32,256)
    float* out = (float*)d_out;                   // (2,64,120,160)

    _Float16* lproj = (_Float16*)d_ws;                          // 240*160*32 f16
    _Float16* rproj = lproj + (size_t)ROWS * Ww * Oc;           // +2.46 MB

    // 2400 pixel tiles / 8 waves per block = 300 blocks; y selects left/right.
    proj_kernel<<<dim3(300, 2), 256, 0, stream>>>(left, right, wdn, lproj, rproj);
    corr_kernel<<<ROWS, 256, 0, stream>>>(lproj, rproj, out);
}
